// GNLayer_13391708029602
// MI455X (gfx1250) — matrix operand order repaired and checked by reading, not compiled, not run on hardware
//
#include <hip/hip_runtime.h>
#include <hip/hip_bf16.h>

#define HDIM 64
#define WAVES 8
#define TPB (WAVES * 32)

typedef __attribute__((ext_vector_type(16))) _Float16 v16h;
typedef __attribute__((ext_vector_type(8)))  float    v8f;

__device__ inline v8f wmma_f16(v16h a, v16h b, v8f c) {
  // D = A(16x32 f16) x B(32x16 f16) + C(16x16 f32)
  return __builtin_amdgcn_wmma_f32_16x16x32_f16(
      false, a, false, b, (short)0, c, false, false);
}

// A fragment (16x32, f16) built from a global f32 row.
// lane<16: K = {kb..kb+7, kb+16..kb+23} with kb=0 ; lane>=16: kb=8.
__device__ inline v16h load_afrag_global(const float* __restrict__ p, int kb) {
  union { v16h v; _Float16 h[16]; } a;
  float t[16];
  *reinterpret_cast<float4*>(&t[0])  = *reinterpret_cast<const float4*>(p + kb);
  *reinterpret_cast<float4*>(&t[4])  = *reinterpret_cast<const float4*>(p + kb + 4);
  *reinterpret_cast<float4*>(&t[8])  = *reinterpret_cast<const float4*>(p + kb + 16);
  *reinterpret_cast<float4*>(&t[12]) = *reinterpret_cast<const float4*>(p + kb + 20);
#pragma unroll
  for (int i = 0; i < 16; ++i) a.h[i] = (_Float16)t[i];
  return a.v;
}

// A fragment from an LDS f16 matrix X[m][k] (row stride K halves).
__device__ inline v16h load_afrag_lds(const _Float16* X, int K, int m, int k0, int kb) {
  union { v16h v; uint4 u[2]; } a;
  const _Float16* p = X + m * K + k0 + kb;
  a.u[0] = *reinterpret_cast<const uint4*>(p);       // K = kb .. kb+7
  a.u[1] = *reinterpret_cast<const uint4*>(p + 16);  // K = kb+16 .. kb+23
  return a.v;
}

// B fragment (32x16, f16) from LDS W^T[n][k] (row stride K halves).
// lane<16: K = {k0..k0+7, k0+16..k0+23} ; lane>=16: K = {k0+8..k0+15, k0+24..k0+31} ; column n = lane&15.
__device__ inline v16h load_bfrag_lds(const _Float16* Wt, int K, int n, int k0, int hi) {
  union { v16h v; uint4 u[2]; } b;
  const _Float16* p = Wt + n * K + k0 + hi * 8;
  b.u[0] = *reinterpret_cast<const uint4*>(p);
  b.u[1] = *reinterpret_cast<const uint4*>(p + 16);
  return b.v;
}

__global__ void gn_zero_kernel(float4* __restrict__ p, long n4) {
  long i = (long)blockIdx.x * blockDim.x + threadIdx.x;
  if (i < n4) p[i] = make_float4(0.f, 0.f, 0.f, 0.f);
}

// ---------------- Edge update: new_edge = MLP([vf[s] | vf[r] | ef]) ; aggr[r] += new_edge
__global__ __launch_bounds__(TPB) void gn_edge_kernel(
    const float* __restrict__ vf, const float* __restrict__ ef,
    const int* __restrict__ eidx,
    const float* __restrict__ We1, const float* __restrict__ be1,
    const float* __restrict__ We2, const float* __restrict__ be2,
    float* __restrict__ out_e, float* __restrict__ aggr,
    long E, int nTiles)
{
  __shared__ __align__(16) _Float16 We1T[HDIM * 192];        // W1^T [n][k], 24 KB
  __shared__ __align__(16) _Float16 We2T[HDIM * HDIM];       // W2^T [n][k],  8 KB
  __shared__ __align__(16) _Float16 h1buf[WAVES][16 * HDIM]; // per-wave 16x64 tile, 16 KB

  const int tid = threadIdx.x;
  for (int i = tid; i < 192 * HDIM; i += TPB) {
    int k = i / HDIM, n = i % HDIM;
    We1T[n * 192 + k] = (_Float16)We1[i];
  }
  for (int i = tid; i < HDIM * HDIM; i += TPB) {
    int k = i / HDIM, n = i % HDIM;
    We2T[n * HDIM + k] = (_Float16)We2[i];
  }
  __syncthreads();

  const int wid  = tid >> 5;
  const int lane = tid & 31;
  const int rrow = lane & 15;   // A-row / B-col / C-col owned by this lane
  const int hi   = lane >> 4;
  const int kb   = hi * 8;

  int  t  = blockIdx.x * WAVES + wid;
  int  tc = t < nTiles ? t : nTiles - 1;
  long e0 = (long)tc * 16;
  long er = e0 + rrow;
  long eS = er < E ? er : E - 1;

  int sidx = eidx[eS];
  int ridx = eidx[E + eS];
  const float* seg0 = vf + (long)sidx * HDIM;  // sender features
  const float* seg1 = vf + (long)ridx * HDIM;  // receiver features
  const float* seg2 = ef + eS * HDIM;          // edge features
  const float* segs[6] = { seg0, seg0 + 32, seg1, seg1 + 32, seg2, seg2 + 32 };

  // ---- GEMM1: [16x192] x [192x64], bias in accumulator
  v8f c[4];
#pragma unroll
  for (int nt = 0; nt < 4; ++nt) {
    float b = be1[nt * 16 + rrow];
#pragma unroll
    for (int j = 0; j < 8; ++j) c[nt][j] = b;
  }
#pragma unroll
  for (int s = 0; s < 6; ++s) {
    v16h a = load_afrag_global(segs[s], kb);
#pragma unroll
    for (int nt = 0; nt < 4; ++nt) {
      v16h b = load_bfrag_lds(We1T, 192, nt * 16 + rrow, s * 32, hi);
      c[nt] = wmma_f16(a, b, c[nt]);
    }
  }

  // ---- ReLU + transpose staging through per-wave LDS (f16)
  _Float16* h1 = &h1buf[wid][0];
#pragma unroll
  for (int nt = 0; nt < 4; ++nt)
#pragma unroll
    for (int j = 0; j < 8; ++j) {
      float v = c[nt][j] > 0.f ? c[nt][j] : 0.f;
      h1[(hi * 8 + j) * HDIM + nt * 16 + rrow] = (_Float16)v;
    }
  __syncthreads();

  // ---- GEMM2: [16x64] x [64x64]
  v8f d[4];
#pragma unroll
  for (int nt = 0; nt < 4; ++nt) {
    float b = be2[nt * 16 + rrow];
#pragma unroll
    for (int j = 0; j < 8; ++j) d[nt][j] = b;
  }
#pragma unroll
  for (int s = 0; s < 2; ++s) {
    v16h a = load_afrag_lds(h1, HDIM, rrow, s * 32, kb);
#pragma unroll
    for (int nt = 0; nt < 4; ++nt) {
      v16h b = load_bfrag_lds(We2T, HDIM, nt * 16 + rrow, s * 32, hi);
      d[nt] = wmma_f16(a, b, d[nt]);
    }
  }

  // receiver index + validity for the 8 rows this lane owns
  int  rIdxRow[8];
  bool rvalid[8];
#pragma unroll
  for (int j = 0; j < 8; ++j) {
    long em = e0 + hi * 8 + j;
    rvalid[j]  = (t < nTiles) && (em < E);
    long emS   = em < E ? em : E - 1;
    rIdxRow[j] = eidx[E + emS];
  }

#pragma unroll
  for (int nt = 0; nt < 4; ++nt) {
    int col = nt * 16 + rrow;
#pragma unroll
    for (int j = 0; j < 8; ++j) {
      if (rvalid[j]) {
        long em = e0 + hi * 8 + j;
        out_e[em * HDIM + col] = d[nt][j];
        __hip_atomic_fetch_add(&aggr[(long)rIdxRow[j] * HDIM + col], d[nt][j],
                               __ATOMIC_RELAXED, __HIP_MEMORY_SCOPE_AGENT);
      }
    }
  }
}

// ---------------- Vertex update: new_vertex = MLP([vf | aggr])
__global__ __launch_bounds__(TPB) void gn_vertex_kernel(
    const float* __restrict__ vf, const float* __restrict__ aggr,
    const float* __restrict__ Wv1, const float* __restrict__ bv1,
    const float* __restrict__ Wv2, const float* __restrict__ bv2,
    float* __restrict__ out_v, long N, int nTiles)
{
  __shared__ __align__(16) _Float16 Wv1T[HDIM * 128];        // 16 KB
  __shared__ __align__(16) _Float16 Wv2T[HDIM * HDIM];       //  8 KB
  __shared__ __align__(16) _Float16 h1buf[WAVES][16 * HDIM]; // 16 KB

  const int tid = threadIdx.x;
  for (int i = tid; i < 128 * HDIM; i += TPB) {
    int k = i / HDIM, n = i % HDIM;
    Wv1T[n * 128 + k] = (_Float16)Wv1[i];
  }
  for (int i = tid; i < HDIM * HDIM; i += TPB) {
    int k = i / HDIM, n = i % HDIM;
    Wv2T[n * HDIM + k] = (_Float16)Wv2[i];
  }
  __syncthreads();

  const int wid  = tid >> 5;
  const int lane = tid & 31;
  const int rrow = lane & 15;
  const int hi   = lane >> 4;
  const int kb   = hi * 8;

  int  t  = blockIdx.x * WAVES + wid;
  int  tc = t < nTiles ? t : nTiles - 1;
  long v0 = (long)tc * 16;
  long vr = v0 + rrow;
  long vS = vr < N ? vr : N - 1;

  const float* seg0 = vf   + vS * HDIM;
  const float* seg1 = aggr + vS * HDIM;
  const float* segs[4] = { seg0, seg0 + 32, seg1, seg1 + 32 };

  v8f c[4];
#pragma unroll
  for (int nt = 0; nt < 4; ++nt) {
    float b = bv1[nt * 16 + rrow];
#pragma unroll
    for (int j = 0; j < 8; ++j) c[nt][j] = b;
  }
#pragma unroll
  for (int s = 0; s < 4; ++s) {
    v16h a = load_afrag_global(segs[s], kb);
#pragma unroll
    for (int nt = 0; nt < 4; ++nt) {
      v16h b = load_bfrag_lds(Wv1T, 128, nt * 16 + rrow, s * 32, hi);
      c[nt] = wmma_f16(a, b, c[nt]);
    }
  }

  _Float16* h1 = &h1buf[wid][0];
#pragma unroll
  for (int nt = 0; nt < 4; ++nt)
#pragma unroll
    for (int j = 0; j < 8; ++j) {
      float v = c[nt][j] > 0.f ? c[nt][j] : 0.f;
      h1[(hi * 8 + j) * HDIM + nt * 16 + rrow] = (_Float16)v;
    }
  __syncthreads();

  v8f d[4];
#pragma unroll
  for (int nt = 0; nt < 4; ++nt) {
    float b = bv2[nt * 16 + rrow];
#pragma unroll
    for (int j = 0; j < 8; ++j) d[nt][j] = b;
  }
#pragma unroll
  for (int s = 0; s < 2; ++s) {
    v16h a = load_afrag_lds(h1, HDIM, rrow, s * 32, kb);
#pragma unroll
    for (int nt = 0; nt < 4; ++nt) {
      v16h b = load_bfrag_lds(Wv2T, HDIM, nt * 16 + rrow, s * 32, hi);
      d[nt] = wmma_f16(a, b, d[nt]);
    }
  }

#pragma unroll
  for (int nt = 0; nt < 4; ++nt) {
    int col = nt * 16 + rrow;
#pragma unroll
    for (int j = 0; j < 8; ++j) {
      long vm = v0 + hi * 8 + j;
      if ((t < nTiles) && (vm < N)) out_v[vm * HDIM + col] = d[nt][j];
    }
  }
}

extern "C" void kernel_launch(void* const* d_in, const int* in_sizes, int n_in,
                              void* d_out, int out_size, void* d_ws, size_t ws_size,
                              hipStream_t stream) {
  const float* vf  = (const float*)d_in[0];
  const float* ef  = (const float*)d_in[1];
  const int*   ei  = (const int*)  d_in[2];
  const float* We1 = (const float*)d_in[3];
  const float* be1 = (const float*)d_in[4];
  const float* We2 = (const float*)d_in[5];
  const float* be2 = (const float*)d_in[6];
  const float* Wv1 = (const float*)d_in[7];
  const float* bv1 = (const float*)d_in[8];
  const float* Wv2 = (const float*)d_in[9];
  const float* bv2 = (const float*)d_in[10];

  long N = in_sizes[0] / HDIM;
  long E = in_sizes[1] / HDIM;

  float* out_v = (float*)d_out;            // new_vertex [N, 64] first
  float* out_e = out_v + N * HDIM;         // new_edge   [E, 64] second
  float* aggr  = (float*)d_ws;             // [N, 64] scatter-sum scratch

  // 1) zero the aggregation buffer
  long n4 = (N * HDIM) / 4;
  gn_zero_kernel<<<(int)((n4 + 255) / 256), 256, 0, stream>>>((float4*)aggr, n4);

  // 2) edge MLP + scatter-sum
  int eTiles = (int)((E + 15) / 16);
  int eBlocks = (eTiles + WAVES - 1) / WAVES;
  gn_edge_kernel<<<eBlocks, TPB, 0, stream>>>(vf, ef, ei, We1, be1, We2, be2,
                                              out_e, aggr, E, eTiles);

  // 3) vertex MLP
  int vTiles = (int)((N + 15) / 16);
  int vBlocks = (vTiles + WAVES - 1) / WAVES;
  gn_vertex_kernel<<<vBlocks, TPB, 0, stream>>>(vf, aggr, Wv1, bv1, Wv2, bv2,
                                                out_v, N, vTiles);
}
